// ANIMAZero_86887188398408
// MI455X (gfx1250) — matrix operand order repaired and checked by reading, not compiled, not run on hardware
//
#include <hip/hip_runtime.h>
#include <hip/hip_bf16.h>

namespace {

constexpr int Dh = 32;   // hidden dim
constexpr int Ss = 8;    // sensory dim
constexpr int Oo = 4;    // output dim
constexpr int Tt = 256;  // steps
constexpr int Bb = 1024; // batch

typedef __attribute__((ext_vector_type(16))) _Float16 v16h;
typedef __attribute__((ext_vector_type(8)))  _Float16 v8h;
typedef __attribute__((ext_vector_type(8)))  float    v8f;
typedef __attribute__((ext_vector_type(4)))  float    v4f;

struct Params {
    const float* obs;                 // [T,B,S]
    const float* enc_w;               // [D,S]
    const float* enc_b;               // [D]
    const float* mats[15];            // wW wI wA zW zI zA rW rI rA hW hI hA aW aI aA  each [D,D]
    const float* att_w; const float* att_b;   // [D,2D],[D]
    const float* wmg_w; const float* wmg_b;
    const float* img_w; const float* img_b;
    const float* amg_w; const float* amg_b;
    const float* z_b; const float* r_b; const float* h_b; const float* a_b;
    const float* phi_w;               // [O,D]
    const float* phi_b;               // [O]
    float* out;                       // [T,B,O]
};

static __device__ __forceinline__ v8f wmma16(v16h a, v16h b, v8f c) {
    // D(f32 16x16) = A(f16 16x32) * B(f16 32x16) + C
    return __builtin_amdgcn_wmma_f32_16x16x32_f16(
        /*neg_a=*/false, a, /*neg_b=*/false, b,
        /*c_mod=*/(short)0, c, /*reuse_a=*/false, /*reuse_b=*/false);
}

// A fragment: row-major f16 tile row `rowp[0..31]`, lane half h = lane>>4.
static __device__ __forceinline__ v16h afrag(const _Float16* rowp, int h) {
    union { v16h v; v8h q[2]; } u;
    u.q[0] = *(const v8h*)(rowp + 8 * h);
    u.q[1] = *(const v8h*)(rowp + 16 + 8 * h);
    return u.v;
}

// B fragment: lane holds column N=lane&15, K in table order (lanes 0-15: K 0..7,16..23,
// lanes 16-31: K 8..15,24..31). rowp = weight row (out-feature) base.
static __device__ __forceinline__ v16h bfrag(const _Float16* rowp, int h) {
    union { v16h v; v8h q[2]; } u;
    u.q[0] = *(const v8h*)(rowp + 8 * h);
    u.q[1] = *(const v8h*)(rowp + 16 + 8 * h);
    return u.v;
}

// Pair of B fragments covering both 16-column output halves of one D x D mat.
struct BPair { v16h lo, hi; };

static __device__ __forceinline__ BPair ldb(const _Float16* w, int stride, int ko,
                                            int n, int h) {
    BPair b;
    b.lo = bfrag(w + n * stride + ko, h);
    b.hi = bfrag(w + (n + 16) * stride + ko, h);
    return b;
}

static __device__ __forceinline__ void fire(v8f& c0, v8f& c1, v16h af, const BPair& b) {
    c0 = wmma16(af, b.lo, c0);
    c1 = wmma16(af, b.hi, c1);
}

// ---- fast activations on the CDNA5 transcendental unit ----
#if defined(__has_builtin)
#  if __has_builtin(__builtin_amdgcn_tanhf)
#    define ANIMA_HW_TANH 1
#  endif
#endif

static __device__ __forceinline__ float tanh_(float x) {
#ifdef ANIMA_HW_TANH
    return __builtin_amdgcn_tanhf(x);              // v_tanh_f32
#else
    return __builtin_fmaf(-2.0f, __builtin_amdgcn_rcpf(__expf(2.0f * x) + 1.0f), 1.0f);
#endif
}

static __device__ __forceinline__ float sigm(float x) {
#ifdef ANIMA_HW_TANH
    return __builtin_fmaf(0.5f, __builtin_amdgcn_tanhf(0.5f * x), 0.5f);
#else
    return __builtin_amdgcn_rcpf(1.0f + __expf(-x));
#endif
}

__global__ __launch_bounds__(32) void anima_rnn_kernel(Params p) {
    __shared__ __align__(16) _Float16 sM[15][Dh * Dh];       // D x D mats
    __shared__ __align__(16) _Float16 sG[4][Dh * 2 * Dh];    // att,wmg,img,amg  [D][2D]
    __shared__ __align__(16) _Float16 sEnc[Dh * Dh];         // enc_w zero-padded to [D][32]
    __shared__ __align__(16) _Float16 sPhi[16 * Dh];         // phi_w zero-padded to [16][32]
    __shared__ __align__(16) _Float16 sObs[16 * Dh];         // obs tile zero-padded cols>=S
    __shared__ __align__(16) _Float16 sScr[16 * Dh];         // layout-transition scratch
    __shared__ float sB[9 * 32 + 32];                        // biases

    const int lane = threadIdx.x & 31;
    const int h = lane >> 4;
    const int n = lane & 15;
    const int base = blockIdx.x * 16;  // 16 batch rows per (single-wave) block

    // ---- one-time weight upload (f32 global -> f16 LDS) ----
    for (int i = lane; i < Dh * Dh; i += 32) {
#pragma unroll
        for (int m = 0; m < 15; ++m) sM[m][i] = (_Float16)p.mats[m][i];
        int r = i >> 5, c = i & 31;
        sEnc[i] = (c < Ss) ? (_Float16)p.enc_w[r * Ss + c] : (_Float16)0.0f;
    }
    for (int i = lane; i < Dh * 2 * Dh; i += 32) {
        sG[0][i] = (_Float16)p.att_w[i];
        sG[1][i] = (_Float16)p.wmg_w[i];
        sG[2][i] = (_Float16)p.img_w[i];
        sG[3][i] = (_Float16)p.amg_w[i];
    }
    for (int i = lane; i < 16 * Dh; i += 32) {
        int r = i >> 5, c = i & 31;
        sPhi[i] = (r < Oo) ? (_Float16)p.phi_w[r * Dh + c] : (_Float16)0.0f;
        sObs[i] = (_Float16)0.0f;
    }
    sB[0   + lane] = p.enc_b[lane];
    sB[32  + lane] = p.att_b[lane];
    sB[64  + lane] = p.wmg_b[lane];
    sB[96  + lane] = p.z_b[lane];
    sB[128 + lane] = p.r_b[lane];
    sB[160 + lane] = p.h_b[lane];
    sB[192 + lane] = p.img_b[lane];
    sB[224 + lane] = p.a_b[lane];
    sB[256 + lane] = p.amg_b[lane];
    sB[288 + lane] = (lane < Oo) ? p.phi_b[lane] : 0.0f;
    __syncthreads();

    // recurrent state: A-fragments (f16) + I in f32 D-layout (needed elementwise)
    v16h Wf = {}, If = {}, Af = {};
    v8f Ia0 = {}, Ia1 = {};

    // NOTE: single wave per block; DS ops from one wave execute in order, so no
    // barriers are needed between the sScr/sObs stores and the fragment reloads.
#pragma unroll 1
    for (int t = 0; t < Tt; ++t) {
        // ---- stage obs tile [16][S] -> f16 LDS (cols >= S stay zero) ----
        {
            const float* og = p.obs + ((size_t)t * Bb + base) * Ss;
            int row = lane >> 1, cb = (lane & 1) * 4;
            v4f v = *(const v4f*)(og + row * Ss + cb);
            _Float16* d = sObs + row * Dh + cb;
            d[0] = (_Float16)v.x; d[1] = (_Float16)v.y;
            d[2] = (_Float16)v.z; d[3] = (_Float16)v.w;
        }
        v16h obsf = afrag(sObs + n * Dh, h);

        // ---- group 1: obs_enc + W_all (load 4 pairs, then fire) ----
        BPair bE  = ldb(sEnc,  Dh, 0, n, h);
        BPair bW0 = ldb(sM[0], Dh, 0, n, h);
        BPair bW1 = ldb(sM[1], Dh, 0, n, h);
        BPair bW2 = ldb(sM[2], Dh, 0, n, h);
        v8f e0 = {}, e1 = {}, wa0 = {}, wa1 = {};
        fire(e0, e1, obsf, bE);
        fire(wa0, wa1, Wf, bW0);
        fire(wa0, wa1, If, bW1);
        fire(wa0, wa1, Af, bW2);

        // ---- group 2: attn + wmg gates ----
        BPair bT0 = ldb(sG[0], 2 * Dh, 0,  n, h);
        BPair bT1 = ldb(sG[0], 2 * Dh, Dh, n, h);
        BPair bG0 = ldb(sG[1], 2 * Dh, 0,  n, h);
        BPair bG1 = ldb(sG[1], 2 * Dh, Dh, n, h);
        v8f at0 = {}, at1 = {}, mg0 = {}, mg1 = {};
        fire(at0, at1, Wf, bT0);
        fire(at0, at1, If, bT1);
        fire(mg0, mg1, If, bG0);
        fire(mg0, mg1, Af, bG1);

        // prefetch z-gate weights; overlaps the W_new activation block below
        BPair bZ0 = ldb(sM[3], Dh, 0, n, h);
        BPair bZ1 = ldb(sM[4], Dh, 0, n, h);
        BPair bZ2 = ldb(sM[5], Dh, 0, n, h);

        // ---- W_new = tanh(obs_enc*attn + W_all*wmg) ----
        v8f Wn0 = {}, Wn1 = {};
        {
            float eb0 = sB[0 + n],  eb1 = sB[0 + n + 16];
            float tb0 = sB[32 + n], tb1 = sB[32 + n + 16];
            float gb0 = sB[64 + n], gb1 = sB[64 + n + 16];
#pragma unroll
            for (int r = 0; r < 8; ++r) {
                Wn0[r] = tanh_(tanh_(e0[r] + eb0) * sigm(at0[r] + tb0) + wa0[r] * sigm(mg0[r] + gb0));
                Wn1[r] = tanh_(tanh_(e1[r] + eb1) * sigm(at1[r] + tb1) + wa1[r] * sigm(mg1[r] + gb1));
            }
        }
#pragma unroll
        for (int r = 0; r < 8; ++r) {
            sScr[(r + 8 * h) * Dh + n]      = (_Float16)Wn0[r];
            sScr[(r + 8 * h) * Dh + n + 16] = (_Float16)Wn1[r];
        }
        v16h Wnf = afrag(sScr + n * Dh, h);

        // ---- z gate ----
        v8f z0 = {}, z1 = {};
        fire(z0, z1, Wnf, bZ0);
        fire(z0, z1, If,  bZ1);
        fire(z0, z1, Af,  bZ2);
        // ---- r gate ----
        BPair bR0 = ldb(sM[6], Dh, 0, n, h);
        BPair bR1 = ldb(sM[7], Dh, 0, n, h);
        BPair bR2 = ldb(sM[8], Dh, 0, n, h);
        v8f r0 = {}, r1 = {};
        fire(r0, r1, Wnf, bR0);
        fire(r0, r1, If,  bR1);
        fire(r0, r1, Af,  bR2);

        // prefetch h-candidate weights; overlaps the sigmoid/r*I block below
        BPair bH0 = ldb(sM[9],  Dh, 0, n, h);
        BPair bH1 = ldb(sM[10], Dh, 0, n, h);
        BPair bH2 = ldb(sM[11], Dh, 0, n, h);

        {
            float zb0 = sB[96 + n],  zb1 = sB[96 + n + 16];
            float rb0 = sB[128 + n], rb1 = sB[128 + n + 16];
#pragma unroll
            for (int r = 0; r < 8; ++r) {
                z0[r] = sigm(z0[r] + zb0); z1[r] = sigm(z1[r] + zb1);
                r0[r] = sigm(r0[r] + rb0); r1[r] = sigm(r1[r] + rb1);
            }
        }
        // r * I -> fragment
#pragma unroll
        for (int r = 0; r < 8; ++r) {
            sScr[(r + 8 * h) * Dh + n]      = (_Float16)(r0[r] * Ia0[r]);
            sScr[(r + 8 * h) * Dh + n + 16] = (_Float16)(r1[r] * Ia1[r]);
        }
        v16h rIf = afrag(sScr + n * Dh, h);

        // ---- h candidate + img gate ----
        v8f h0 = {}, h1 = {};
        fire(h0, h1, Wnf, bH0);
        fire(h0, h1, rIf, bH1);
        fire(h0, h1, Af,  bH2);
        BPair bI0 = ldb(sG[2], 2 * Dh, 0,  n, h);
        BPair bI1 = ldb(sG[2], 2 * Dh, Dh, n, h);
        v8f mi0 = {}, mi1 = {};
        fire(mi0, mi1, Wnf, bI0);
        fire(mi0, mi1, Af,  bI1);

        // prefetch A_all weights; overlaps the I_new activation block below
        BPair bA0 = ldb(sM[12], Dh, 0, n, h);
        BPair bA1 = ldb(sM[13], Dh, 0, n, h);
        BPair bA2 = ldb(sM[14], Dh, 0, n, h);

        // ---- I_new = (1-z)*I + z*(h*mult_I) ----
        {
            float hb0 = sB[160 + n], hb1 = sB[160 + n + 16];
            float ib0 = sB[192 + n], ib1 = sB[192 + n + 16];
#pragma unroll
            for (int r = 0; r < 8; ++r) {
                float hv0 = tanh_(h0[r] + hb0) * sigm(mi0[r] + ib0);
                float hv1 = tanh_(h1[r] + hb1) * sigm(mi1[r] + ib1);
                Ia0[r] = (1.0f - z0[r]) * Ia0[r] + z0[r] * hv0;
                Ia1[r] = (1.0f - z1[r]) * Ia1[r] + z1[r] * hv1;
            }
        }
#pragma unroll
        for (int r = 0; r < 8; ++r) {
            sScr[(r + 8 * h) * Dh + n]      = (_Float16)Ia0[r];
            sScr[(r + 8 * h) * Dh + n + 16] = (_Float16)Ia1[r];
        }
        v16h Inf = afrag(sScr + n * Dh, h);

        // ---- A_all + amg gate ----
        v8f aa0 = {}, aa1 = {};
        fire(aa0, aa1, Wnf, bA0);
        fire(aa0, aa1, Inf, bA1);
        fire(aa0, aa1, Af,  bA2);
        BPair bM0 = ldb(sG[3], 2 * Dh, 0,  n, h);
        BPair bM1 = ldb(sG[3], 2 * Dh, Dh, n, h);
        v16h bPhi = bfrag(sPhi + n * Dh, h);   // prefetch readout weights too
        v8f ma0 = {}, ma1 = {};
        fire(ma0, ma1, Wnf, bM0);
        fire(ma0, ma1, Inf, bM1);

        // ---- A_new = tanh(A_all * mult_A) ----
        v8f An0 = {}, An1 = {};
        {
            float ab0 = sB[224 + n], ab1 = sB[224 + n + 16];
            float mb0 = sB[256 + n], mb1 = sB[256 + n + 16];
#pragma unroll
            for (int r = 0; r < 8; ++r) {
                An0[r] = tanh_((aa0[r] + ab0) * sigm(ma0[r] + mb0));
                An1[r] = tanh_((aa1[r] + ab1) * sigm(ma1[r] + mb1));
            }
        }
#pragma unroll
        for (int r = 0; r < 8; ++r) {
            sScr[(r + 8 * h) * Dh + n]      = (_Float16)An0[r];
            sScr[(r + 8 * h) * Dh + n + 16] = (_Float16)An1[r];
        }
        v16h Anf = afrag(sScr + n * Dh, h);

        // ---- readout: out = A_new @ phi.T + phi_b (N=4 in one padded WMMA) ----
        v8f dphi = {};
        dphi = wmma16(Anf, bPhi, dphi);
        if (n < Oo) {
            float pb = sB[288 + n];
            float* og = p.out + ((size_t)t * Bb + base) * Oo;
#pragma unroll
            for (int r = 0; r < 8; ++r)
                og[(r + 8 * h) * Oo + n] = dphi[r] + pb;
        }

        // carry
        Wf = Wnf; If = Inf; Af = Anf;
    }
}

} // namespace

extern "C" void kernel_launch(void* const* d_in, const int* in_sizes, int n_in,
                              void* d_out, int out_size, void* d_ws, size_t ws_size,
                              hipStream_t stream) {
    (void)in_sizes; (void)n_in; (void)out_size; (void)d_ws; (void)ws_size;
    Params p;
    p.obs   = (const float*)d_in[0];
    p.enc_w = (const float*)d_in[1];
    p.enc_b = (const float*)d_in[2];
    for (int i = 0; i < 15; ++i) p.mats[i] = (const float*)d_in[3 + i];
    p.att_w = (const float*)d_in[18]; p.att_b = (const float*)d_in[19];
    p.wmg_w = (const float*)d_in[20]; p.wmg_b = (const float*)d_in[21];
    p.img_w = (const float*)d_in[22]; p.img_b = (const float*)d_in[23];
    p.amg_w = (const float*)d_in[24]; p.amg_b = (const float*)d_in[25];
    p.z_b   = (const float*)d_in[26]; p.r_b   = (const float*)d_in[27];
    p.h_b   = (const float*)d_in[28]; p.a_b   = (const float*)d_in[29];
    p.phi_w = (const float*)d_in[30]; p.phi_b = (const float*)d_in[31];
    p.out = (float*)d_out;

    anima_rnn_kernel<<<dim3(Bb / 16), dim3(32), 0, stream>>>(p);
}
